// __RelativePositionalMultiheadAttentionPT_73117523247464
// MI455X (gfx1250) — matrix operand order repaired and checked by reading, not compiled, not run on hardware
//
#include <hip/hip_runtime.h>

#define NUM_HEADS 12
#define HD 64
#define LQ 1024
#define CDIM 768
#define BATCH 8
#define BHD (BATCH * NUM_HEADS) // 96

typedef __attribute__((ext_vector_type(16))) __bf16 v16bf;
typedef __attribute__((ext_vector_type(8)))  float  v8f;

union BfFrag {
  uint4 u[2];
  v16bf v;
};
static_assert(sizeof(v16bf) == 32, "v16bf size");

__device__ __forceinline__ unsigned short f2bf(float f) {
  unsigned int u = __float_as_uint(f);
  u += 0x7FFFu + ((u >> 16) & 1u);          // round-to-nearest-even
  return (unsigned short)(u >> 16);
}
__device__ __forceinline__ float bf2f(unsigned short h) {
  return __uint_as_float(((unsigned int)h) << 16);
}
__device__ __forceinline__ v8f wmma_bf16(v16bf a, v16bf b, v8f c) {
  return __builtin_amdgcn_wmma_f32_16x16x32_bf16(
      /*neg_a=*/false, a, /*neg_b=*/false, b,
      /*c_mod=*/(short)0, c, /*reuse_a=*/false, /*reuse_b=*/false);
}

// Async DMA: global (16B per lane) -> LDS, tracked by ASYNCcnt.
// Low 32 bits of a generic pointer to LDS are the LDS byte offset.
__device__ __forceinline__ void async_ld_b128(const unsigned short* gptr,
                                              unsigned short* lptr) {
  unsigned lds = (unsigned)(unsigned long long)lptr;
  asm volatile("global_load_async_to_lds_b128 %0, %1, off"
               :: "v"(lds), "v"(gptr)
               : "memory");
}

// ---------------------------------------------------------------------------
// Kernel 1: QKV projection GEMM  (8192 x 768) @ (768 x 2304) + bias
// Epilogue scatters bf16 into Q (bh,l,d), K (bh,l,d), Vt (bh,d,l).
// Block: 256 threads = 8 waves (2 along M x 4 along N), tile 128x128, BK=32.
// ---------------------------------------------------------------------------
__global__ __launch_bounds__(256) void qkv_gemm_kernel(
    const float* __restrict__ X, const float* __restrict__ W,
    const float* __restrict__ bias,
    unsigned short* __restrict__ Qb, unsigned short* __restrict__ Kb,
    unsigned short* __restrict__ Vt)
{
  __shared__ __align__(16) unsigned short shA[128 * 40]; // (m,k) padded 40
  __shared__ __align__(16) unsigned short shB[128 * 40]; // (n,k) padded 40
  const int t = threadIdx.x;
  const int n0 = blockIdx.x * 128;
  const int m0 = blockIdx.y * 128;
  const int wave = t >> 5, lane = t & 31;
  const int hl = lane >> 4, nl = lane & 15;
  const int wm = wave & 1, wn = wave >> 1;

  v8f acc[4][2] = {};

  for (int k0 = 0; k0 < CDIM; k0 += 32) {
    __syncthreads();
    // Stage A: 128x32 fp32 -> bf16
    #pragma unroll
    for (int i = 0; i < 4; ++i) {
      int cidx = i * 256 + t;                 // 1024 float4 chunks
      int row = cidx >> 3, c4 = (cidx & 7) << 2;
      const float* src = X + (size_t)(m0 + row) * CDIM + k0 + c4;
      float4 f = *(const float4*)src;
      if (k0 + 32 < CDIM) __builtin_prefetch(src + 32, 0, 0);
      ushort4 u;
      u.x = f2bf(f.x); u.y = f2bf(f.y); u.z = f2bf(f.z); u.w = f2bf(f.w);
      *(ushort4*)&shA[row * 40 + c4] = u;
    }
    // Stage B transposed: W rows (k) read coalesced over n, write shB[n][k]
    #pragma unroll
    for (int i = 0; i < 4; ++i) {
      int cidx = i * 256 + t;                 // 1024 float4 chunks over 32x128
      int kk = cidx >> 5, nn4 = (cidx & 31) << 2;
      const float* src = W + (size_t)(k0 + kk) * 2304 + n0 + nn4;
      float4 f = *(const float4*)src;
      if (k0 + 32 < CDIM) __builtin_prefetch(src + (size_t)32 * 2304, 0, 0);
      shB[(nn4 + 0) * 40 + kk] = f2bf(f.x);
      shB[(nn4 + 1) * 40 + kk] = f2bf(f.y);
      shB[(nn4 + 2) * 40 + kk] = f2bf(f.z);
      shB[(nn4 + 3) * 40 + kk] = f2bf(f.w);
    }
    __syncthreads();

    BfFrag af[4], bfg[2];
    #pragma unroll
    for (int mi = 0; mi < 4; ++mi) {          // A-frag: two 16B chunks
      const unsigned short* ar = &shA[(wm * 64 + mi * 16 + nl) * 40];
      af[mi].u[0] = *(const uint4*)(ar + hl * 8);
      af[mi].u[1] = *(const uint4*)(ar + 16 + hl * 8);
    }
    #pragma unroll
    for (int ni = 0; ni < 2; ++ni) {          // B-frag: two 16B chunks
      const unsigned short* br = &shB[(wn * 32 + ni * 16 + nl) * 40 + hl * 8];
      bfg[ni].u[0] = *(const uint4*)(br);
      bfg[ni].u[1] = *(const uint4*)(br + 16);
    }
    #pragma unroll
    for (int mi = 0; mi < 4; ++mi)
      #pragma unroll
      for (int ni = 0; ni < 2; ++ni)
        acc[mi][ni] = wmma_bf16(af[mi].v, bfg[ni].v, acc[mi][ni]);
  }

  // Epilogue: add bias, scatter to Q / K / V^T
  #pragma unroll
  for (int mi = 0; mi < 4; ++mi) {
    #pragma unroll
    for (int ni = 0; ni < 2; ++ni) {
      int n = n0 + wn * 32 + ni * 16 + nl;    // global column in 3C
      float bv = bias[n];
      int which = n / CDIM;                   // 0=q 1=k 2=v (uniform per frag)
      int c = n - which * CDIM;
      int head = c >> 6, d = c & 63;
      #pragma unroll
      for (int r = 0; r < 8; ++r) {
        int m = m0 + wm * 64 + mi * 16 + hl * 8 + r;
        int b = m >> 10, lq = m & 1023;
        int bh = b * NUM_HEADS + head;
        unsigned short val = f2bf(acc[mi][ni][r] + bv);
        if (which == 0)
          Qb[((size_t)bh * LQ + lq) * HD + d] = val;
        else if (which == 1)
          Kb[((size_t)bh * LQ + lq) * HD + d] = val;
        else
          Vt[((size_t)bh * HD + d) * LQ + lq] = val;
      }
    }
  }
}

// ---------------------------------------------------------------------------
// Kernel 2: decomposed rel-pos tables
//   relH[bh,q,kh] = dot(q[bh,q,:], rel_pos_h[qh-kh+31,:])
//   relW[bh,q,kw] = dot(q[bh,q,:], rel_pos_w[qw-kw+31,:])
// One block per (bh,q), 64 threads (32 for relH cols, 32 for relW cols).
// ---------------------------------------------------------------------------
__global__ __launch_bounds__(64) void relpos_kernel(
    const unsigned short* __restrict__ Qb,
    const float* __restrict__ rel_pos_h, const float* __restrict__ rel_pos_w,
    float* __restrict__ relH, float* __restrict__ relW)
{
  __shared__ float qrow[HD];
  const int bq = blockIdx.x;                  // bh*1024 + q
  const int t = threadIdx.x;
  qrow[t] = bf2f(Qb[(size_t)bq * HD + t]);
  __syncthreads();
  const int q = bq & (LQ - 1);
  const int qh = q >> 5, qw = q & 31;
  const int col = t & 31;
  const float* rp = (t < 32) ? (rel_pos_h + (size_t)(qh - col + 31) * HD)
                             : (rel_pos_w + (size_t)(qw - col + 31) * HD);
  float s = 0.f;
  #pragma unroll
  for (int d = 0; d < HD; ++d) s += qrow[d] * rp[d];
  if (t < 32) relH[(size_t)bq * 32 + col] = s;
  else        relW[(size_t)bq * 32 + col] = s;
}

// ---------------------------------------------------------------------------
// Kernel 3: fused flash attention with decomposed rel-pos bias.
// Block = 256 threads = 8 waves x 16 queries = 128 queries; all waves share
// the same key tile, cooperatively staged to LDS by async global->LDS DMA,
// double-buffered (issue kt+1, s_wait_asynccnt 2 => tile kt has landed).
// 8 WMMA per 32-key tile: 4 for QK^T, 4 for PV.
// ---------------------------------------------------------------------------
__global__ __launch_bounds__(256) void attn_kernel(
    const unsigned short* __restrict__ Q, const unsigned short* __restrict__ K,
    const unsigned short* __restrict__ Vt,
    const float* __restrict__ relH, const float* __restrict__ relW,
    unsigned short* __restrict__ AttnOut)
{
  __shared__ __align__(16) unsigned short shK[2][32 * 72]; // (key,d) pad 72
  __shared__ __align__(16) unsigned short shV[2][64 * 40]; // (d,key) pad 40
  __shared__ __align__(16) unsigned short pbuf[8][16 * 32];
  const int bh = blockIdx.y;
  const int t = threadIdx.x;
  const int wave = t >> 5, lane = t & 31;
  const int hl = lane >> 4, nl = lane & 15;
  const int qbase = blockIdx.x * 128 + wave * 16;
  unsigned short* myp = pbuf[wave];

  const unsigned short* Qp = Q  + (size_t)bh * LQ * HD;
  const unsigned short* Kp = K  + (size_t)bh * LQ * HD;
  const unsigned short* Vp = Vt + (size_t)bh * HD * LQ;
  const float* rH = relH + (size_t)bh * LQ * 32;
  const float* rW = relW + (size_t)bh * LQ * 32;

  // Per-thread fixed staging coordinates (16B chunks)
  const int krow = t >> 3, kc = (t & 7) * 8;  // K: 32 rows x 8 chunks
  const int vrow = t >> 2, vc = (t & 3) * 8;  // V: 64 rows x 4 chunks

  auto stage = [&](int kt, int buf) {
    const int keyb = kt * 32;
    async_ld_b128(Kp + (size_t)(keyb + krow) * HD + kc,
                  &shK[buf][krow * 72 + kc]);
    async_ld_b128(Vp + (size_t)vrow * LQ + keyb + vc,
                  &shV[buf][vrow * 40 + vc]);
  };
  stage(0, 0);

  // Q A-frags for both 32-deep k-slices of hd=64 (held for the whole loop)
  BfFrag qa[2];
  {
    const unsigned short* qr = Qp + (size_t)(qbase + nl) * HD;
    #pragma unroll
    for (int s = 0; s < 2; ++s) {
      qa[s].u[0] = *(const uint4*)(qr + s * 32 + hl * 8);
      qa[s].u[1] = *(const uint4*)(qr + s * 32 + hl * 8 + 16);
    }
  }

  // rel_w bias is loop-invariant per (row, column-in-tile): preload
  float rw[2][8];
  #pragma unroll
  for (int tt = 0; tt < 2; ++tt)
    #pragma unroll
    for (int r = 0; r < 8; ++r)
      rw[tt][r] = rW[(size_t)(qbase + hl * 8 + r) * 32 + tt * 16 + nl];

  float mrun[8], lrun[8];
  v8f o[4] = {};
  #pragma unroll
  for (int r = 0; r < 8; ++r) { mrun[r] = -1e30f; lrun[r] = 0.f; }

  const float scale = 0.125f; // hd^-0.5 = 1/8

  for (int kt = 0; kt < 32; ++kt) {
    const int cur = kt & 1;
    if (kt + 1 < 32) {
      stage(kt + 1, cur ^ 1);                          // overlap DMA w/ math
      asm volatile("s_wait_asynccnt 0x2" ::: "memory"); // tile kt has landed
    } else {
      asm volatile("s_wait_asynccnt 0x0" ::: "memory");
    }
    __syncthreads();

    // S = Q * K^T : 16 queries x 32 keys (two 16-key N subtiles) from LDS
    v8f s0 = {}, s1 = {};
    {
      const unsigned short* kb0 = &shK[cur][nl * 72];
      const unsigned short* kb1 = &shK[cur][(16 + nl) * 72];
      BfFrag b0;
      b0.u[0] = *(const uint4*)(kb0 + hl * 8);
      b0.u[1] = *(const uint4*)(kb0 + hl * 8 + 16);
      s0 = wmma_bf16(qa[0].v, b0.v, s0);
      b0.u[0] = *(const uint4*)(kb0 + 32 + hl * 8);
      b0.u[1] = *(const uint4*)(kb0 + 32 + hl * 8 + 16);
      s0 = wmma_bf16(qa[1].v, b0.v, s0);
      b0.u[0] = *(const uint4*)(kb1 + hl * 8);
      b0.u[1] = *(const uint4*)(kb1 + hl * 8 + 16);
      s1 = wmma_bf16(qa[0].v, b0.v, s1);
      b0.u[0] = *(const uint4*)(kb1 + 32 + hl * 8);
      b0.u[1] = *(const uint4*)(kb1 + 32 + hl * 8 + 16);
      s1 = wmma_bf16(qa[1].v, b0.v, s1);
    }
    // scale + bias: key_h == kt for the whole tile, key_w == column index
    #pragma unroll
    for (int r = 0; r < 8; ++r) {
      float rh = rH[(size_t)(qbase + hl * 8 + r) * 32 + kt];
      s0[r] = s0[r] * scale + rh + rw[0][r];
      s1[r] = s1[r] * scale + rh + rw[1][r];
    }
    // online softmax (rows live across 16 lanes of one half-wave)
    #pragma unroll
    for (int r = 0; r < 8; ++r) {
      float mx = fmaxf(s0[r], s1[r]);
      mx = fmaxf(mx, __shfl_xor(mx, 1, 32));
      mx = fmaxf(mx, __shfl_xor(mx, 2, 32));
      mx = fmaxf(mx, __shfl_xor(mx, 4, 32));
      mx = fmaxf(mx, __shfl_xor(mx, 8, 32));
      float mn = fmaxf(mrun[r], mx);
      float corr = __expf(mrun[r] - mn);
      mrun[r] = mn;
      s0[r] = __expf(s0[r] - mn);
      s1[r] = __expf(s1[r] - mn);
      float sm = s0[r] + s1[r];
      sm += __shfl_xor(sm, 1, 32);
      sm += __shfl_xor(sm, 2, 32);
      sm += __shfl_xor(sm, 4, 32);
      sm += __shfl_xor(sm, 8, 32);
      lrun[r] = lrun[r] * corr + sm;
      #pragma unroll
      for (int tt = 0; tt < 4; ++tt) o[tt][r] *= corr;
    }
    // P -> LDS bf16 (C-layout to A-layout bounce, wave-local)
    #pragma unroll
    for (int r = 0; r < 8; ++r) {
      myp[(hl * 8 + r) * 32 + nl]      = f2bf(s0[r]);
      myp[(hl * 8 + r) * 32 + 16 + nl] = f2bf(s1[r]);
    }
    BfFrag pa;
    const unsigned short* pr = myp + nl * 32;
    pa.u[0] = *(const uint4*)(pr + hl * 8);
    pa.u[1] = *(const uint4*)(pr + 16 + hl * 8);
    // O += P * V : N = hd = 64 => four 16-wide n-subtiles from LDS V^T
    #pragma unroll
    for (int tt = 0; tt < 4; ++tt) {
      BfFrag vb;
      const unsigned short* vr = &shV[cur][(tt * 16 + nl) * 40 + hl * 8];
      vb.u[0] = *(const uint4*)(vr);
      vb.u[1] = *(const uint4*)(vr + 16);
      o[tt] = wmma_bf16(pa.v, vb.v, o[tt]);
    }
    __syncthreads();   // everyone done with buffer `cur` before it is reused
  }

  // finalize: divide by softmax denom, store to (B, L, C) bf16
  const int b = bh / NUM_HEADS, head = bh - b * NUM_HEADS;
  #pragma unroll
  for (int r = 0; r < 8; ++r) {
    const float inv = 1.f / lrun[r];
    const int qm = qbase + hl * 8 + r;
    #pragma unroll
    for (int tt = 0; tt < 4; ++tt) {
      size_t addr = ((size_t)(b * LQ + qm)) * CDIM + head * HD + tt * 16 + nl;
      AttnOut[addr] = f2bf(o[tt][r] * inv);
    }
  }
}

// ---------------------------------------------------------------------------
// Kernel 4: output projection  (8192 x 768) bf16 @ (768 x 768) + bias -> fp32
// ---------------------------------------------------------------------------
__global__ __launch_bounds__(256) void proj_gemm_kernel(
    const unsigned short* __restrict__ A, const float* __restrict__ W,
    const float* __restrict__ bias, float* __restrict__ out)
{
  __shared__ __align__(16) unsigned short shA[128 * 40];
  __shared__ __align__(16) unsigned short shB[128 * 40];
  const int t = threadIdx.x;
  const int n0 = blockIdx.x * 128;
  const int m0 = blockIdx.y * 128;
  const int wave = t >> 5, lane = t & 31;
  const int hl = lane >> 4, nl = lane & 15;
  const int wm = wave & 1, wn = wave >> 1;

  v8f acc[4][2] = {};

  for (int k0 = 0; k0 < CDIM; k0 += 32) {
    __syncthreads();
    #pragma unroll
    for (int i = 0; i < 4; ++i) {
      int cidx = i * 256 + t;
      int row = cidx >> 3, c4 = (cidx & 7) << 2;
      const unsigned short* src = A + (size_t)(m0 + row) * CDIM + k0 + c4;
      ushort4 u = *(const ushort4*)src;
      if (k0 + 32 < CDIM) __builtin_prefetch(src + 32, 0, 0);
      *(ushort4*)&shA[row * 40 + c4] = u;
    }
    #pragma unroll
    for (int i = 0; i < 4; ++i) {
      int cidx = i * 256 + t;
      int kk = cidx >> 5, nn4 = (cidx & 31) << 2;
      const float* src = W + (size_t)(k0 + kk) * CDIM + n0 + nn4;
      float4 f = *(const float4*)src;
      if (k0 + 32 < CDIM) __builtin_prefetch(src + (size_t)32 * CDIM, 0, 0);
      shB[(nn4 + 0) * 40 + kk] = f2bf(f.x);
      shB[(nn4 + 1) * 40 + kk] = f2bf(f.y);
      shB[(nn4 + 2) * 40 + kk] = f2bf(f.z);
      shB[(nn4 + 3) * 40 + kk] = f2bf(f.w);
    }
    __syncthreads();

    BfFrag af[4], bfg[2];
    #pragma unroll
    for (int mi = 0; mi < 4; ++mi) {
      const unsigned short* ar = &shA[(wm * 64 + mi * 16 + nl) * 40];
      af[mi].u[0] = *(const uint4*)(ar + hl * 8);
      af[mi].u[1] = *(const uint4*)(ar + 16 + hl * 8);
    }
    #pragma unroll
    for (int ni = 0; ni < 2; ++ni) {
      const unsigned short* br = &shB[(wn * 32 + ni * 16 + nl) * 40 + hl * 8];
      bfg[ni].u[0] = *(const uint4*)(br);
      bfg[ni].u[1] = *(const uint4*)(br + 16);
    }
    #pragma unroll
    for (int mi = 0; mi < 4; ++mi)
      #pragma unroll
      for (int ni = 0; ni < 2; ++ni)
        acc[mi][ni] = wmma_bf16(af[mi].v, bfg[ni].v, acc[mi][ni]);
  }

  #pragma unroll
  for (int mi = 0; mi < 4; ++mi)
    #pragma unroll
    for (int ni = 0; ni < 2; ++ni) {
      int n = n0 + wn * 32 + ni * 16 + nl;
      float bv = bias[n];
      #pragma unroll
      for (int r = 0; r < 8; ++r) {
        int m = m0 + wm * 64 + mi * 16 + hl * 8 + r;
        out[(size_t)m * CDIM + n] = acc[mi][ni][r] + bv;
      }
    }
}

// ---------------------------------------------------------------------------
extern "C" void kernel_launch(void* const* d_in, const int* in_sizes, int n_in,
                              void* d_out, int out_size, void* d_ws, size_t ws_size,
                              hipStream_t stream) {
  (void)in_sizes; (void)n_in; (void)out_size; (void)ws_size;
  const float* x      = (const float*)d_in[0];
  const float* qkv_w  = (const float*)d_in[1];
  const float* qkv_b  = (const float*)d_in[2];
  const float* proj_w = (const float*)d_in[3];
  const float* proj_b = (const float*)d_in[4];
  const float* rph    = (const float*)d_in[5];
  const float* rpw    = (const float*)d_in[6];

  // Workspace: Qb,Kb,Vt (bf16) + relH,relW (f32) + AttnOut (bf16) = 75.5 MB
  const size_t SZ = (size_t)BHD * LQ * HD;
  unsigned short* Qb = (unsigned short*)d_ws;
  unsigned short* Kb = Qb + SZ;
  unsigned short* Vt = Kb + SZ;
  float* relH = (float*)(Vt + SZ);
  float* relW = relH + (size_t)BHD * LQ * 32;
  unsigned short* AttnOut = (unsigned short*)(relW + (size_t)BHD * LQ * 32);
  float* out = (float*)d_out;

  qkv_gemm_kernel<<<dim3(2304 / 128, 8192 / 128), 256, 0, stream>>>(
      x, qkv_w, qkv_b, Qb, Kb, Vt);
  relpos_kernel<<<dim3(BHD * LQ), 64, 0, stream>>>(Qb, rph, rpw, relH, relW);
  attn_kernel<<<dim3(LQ / 128, BHD), 256, 0, stream>>>(
      Qb, Kb, Vt, relH, relW, AttnOut);
  proj_gemm_kernel<<<dim3(CDIM / 128, 8192 / 128), 256, 0, stream>>>(
      AttnOut, proj_w, proj_b, out);
}
